// ObjectSlotEncoder_65412351918169
// MI455X (gfx1250) — matrix operand order repaired and checked by reading, not compiled, not run on hardware
//
#include <hip/hip_runtime.h>
#include <hip/hip_bf16.h>

// ---------------------------------------------------------------------------
// Types for CDNA5 WMMA (wave32, 16x16x32 bf16 -> f32)
// ---------------------------------------------------------------------------
typedef __bf16 bf16_t;
typedef __attribute__((ext_vector_type(16))) __bf16       bf16x16;
typedef __attribute__((ext_vector_type(8)))  float        f32x8;
typedef __attribute__((ext_vector_type(4)))  unsigned int u32x4;

#define WMMA_BF16(a, b, c) \
    __builtin_amdgcn_wmma_f32_16x16x32_bf16(false, (a), false, (b), (short)0, (c), false, false)

// ---------------------------------------------------------------------------
// Fragment loaders (per ISA 7.12.2 VGPR layouts, wave32).
//
// A (16x32, MxK, row-major source, stride ld):
//   lane&15 = row M; half = lane>>4; element e -> K = (e&7) + ((e>>3)<<4) + half*8
//   => two contiguous 16B loads per lane at  row*ld + half*8  and +16 elements.
//
// B (32x16, KxN) consumed from "Bt" layout = B transposed, i.e. [N][K] row-major:
//   lane&15 = col N; element e -> K = (e&7) + ((e>>3)<<4) + half*8
//   => two contiguous 16B loads per lane at  col*ld + half*8  and +16 elements.
//
// C/D (16x16 f32): element r -> M = r + half*8 ; N = lane&15.
// ---------------------------------------------------------------------------
__device__ __forceinline__ bf16x16 load_a16(const bf16_t* base, int ld) {
    const int lane    = threadIdx.x & 31;
    const int halfsel = lane >> 4;
    const bf16_t* p = base + (lane & 15) * ld + (halfsel << 3);
    union { u32x4 q[2]; bf16x16 v; } u;
    u.q[0] = *(const u32x4*)(p);
    u.q[1] = *(const u32x4*)(p + 16);
    return u.v;
}

__device__ __forceinline__ bf16x16 load_b16(const bf16_t* base, int ld) {
    const int lane    = threadIdx.x & 31;
    const int halfsel = lane >> 4;
    const bf16_t* p = base + (lane & 15) * ld + (halfsel << 3);
    union { u32x4 q[2]; bf16x16 v; } u;
    u.q[0] = *(const u32x4*)(p);
    u.q[1] = *(const u32x4*)(p + 16);
    return u.v;
}

__device__ __forceinline__ float sigmoidf_(float x) { return 1.0f / (1.0f + __expf(-x)); }

// ---------------------------------------------------------------------------
// Kernel 0: weight cast/transpose to bf16 "Bt" ([N][K]) layout.
// transpose=1: dst[n*K+k] = W[k*N+n] ; transpose=0: plain cast copy.
// ---------------------------------------------------------------------------
__global__ void pack_bt(const float* __restrict__ W, bf16_t* __restrict__ dst,
                        int K, int N, int transpose) {
    int total = K * N;
    for (int i = blockIdx.x * blockDim.x + threadIdx.x; i < total; i += gridDim.x * blockDim.x) {
        if (transpose) {
            int n = i / K, k = i - n * K;
            dst[i] = (bf16_t)W[k * N + n];
        } else {
            dst[i] = (bf16_t)W[i];
        }
    }
}

// ---------------------------------------------------------------------------
// Kernel 1: LayerNorm(features) then fused k/v projection GEMM.
//   grid = 4096 blocks, each handles 16 rows (of 65536) x K=512 -> N=512 (k|v).
//   k stored row-major bf16 [row][256]; v stored transposed bf16 [bt][d][n].
// ---------------------------------------------------------------------------
__global__ __launch_bounds__(256)
void ln_kv_kernel(const float* __restrict__ x,
                  const float* __restrict__ g, const float* __restrict__ beta,
                  const bf16_t* __restrict__ WkvT,
                  const float* __restrict__ bk, const float* __restrict__ bv,
                  bf16_t* __restrict__ kbf, bf16_t* __restrict__ vT) {
    __shared__ __align__(32) float  xf[16 * 512];
    __shared__ __align__(32) bf16_t abf[16 * 512];

    const int tid = threadIdx.x;
    const size_t base = (size_t)blockIdx.x * 16 * 512;

    for (int i = tid; i < 8192; i += 256) xf[i] = x[base + i];
    __syncthreads();

    // LayerNorm: 16 lanes per row, shfl-xor reduction within 16-lane halves (wave32)
    {
        const int row = tid >> 4, l16 = tid & 15;
        float s = 0.f, ss = 0.f;
        #pragma unroll
        for (int j = 0; j < 32; ++j) {
            float v = xf[row * 512 + l16 + 16 * j];
            s += v; ss += v * v;
        }
        #pragma unroll
        for (int m = 1; m < 16; m <<= 1) {
            s  += __shfl_xor(s,  m, 32);
            ss += __shfl_xor(ss, m, 32);
        }
        float mean = s * (1.f / 512.f);
        float var  = ss * (1.f / 512.f) - mean * mean;
        float rstd = rsqrtf(var + 1e-5f);
        #pragma unroll
        for (int j = 0; j < 32; ++j) {
            int c = l16 + 16 * j;
            abf[row * 512 + c] = (bf16_t)((xf[row * 512 + c] - mean) * rstd * g[c] + beta[c]);
        }
    }
    __syncthreads();

    const int wave = tid >> 5, lane = tid & 31;
    const int halfsel = lane >> 4, col16 = lane & 15;

    f32x8 acc[4] = {{}, {}, {}, {}};
    for (int k0 = 0; k0 < 512; k0 += 32) {
        bf16x16 a = load_a16(abf + k0, 512);
        #pragma unroll
        for (int jj = 0; jj < 4; ++jj) {
            int nt = wave + 8 * jj;
            bf16x16 b = load_b16(WkvT + (size_t)nt * 16 * 512 + k0, 512);
            acc[jj] = WMMA_BF16(a, b, acc[jj]);
        }
    }

    #pragma unroll
    for (int jj = 0; jj < 4; ++jj) {
        int n = (wave + 8 * jj) * 16 + col16;
        #pragma unroll
        for (int r = 0; r < 8; ++r) {
            size_t rowg = (size_t)blockIdx.x * 16 + r + halfsel * 8;
            if (n < 256) {
                kbf[rowg * 256 + n] = (bf16_t)(acc[jj][r] + bk[n]);
            } else {
                int d = n - 256;
                size_t bt = rowg >> 10, nl = rowg & 1023;
                vT[((bt * 256 + d) << 10) + nl] = (bf16_t)(acc[jj][r] + bv[d]);
            }
        }
    }
}

// ---------------------------------------------------------------------------
// Helper: LayerNorm of a 16x256 fp32 LDS tile -> bf16 LDS tile.
// Caller provides __syncthreads() fencing.
// ---------------------------------------------------------------------------
__device__ __forceinline__ void ln16(const float* __restrict__ x,
                                     const float* __restrict__ g,
                                     const float* __restrict__ beta,
                                     bf16_t* __restrict__ dst) {
    const int tid = threadIdx.x;
    const int row = tid >> 4, l16 = tid & 15;
    float s = 0.f, ss = 0.f;
    #pragma unroll
    for (int j = 0; j < 16; ++j) {
        float v = x[row * 256 + l16 + 16 * j];
        s += v; ss += v * v;
    }
    #pragma unroll
    for (int m = 1; m < 16; m <<= 1) {
        s  += __shfl_xor(s,  m, 32);
        ss += __shfl_xor(ss, m, 32);
    }
    float mean = s * (1.f / 256.f);
    float var  = ss * (1.f / 256.f) - mean * mean;
    float rstd = rsqrtf(var + 1e-5f);
    #pragma unroll
    for (int j = 0; j < 16; ++j) {
        int c = l16 + 16 * j;
        dst[row * 256 + c] = (bf16_t)((x[row * 256 + c] - mean) * rstd * g[c] + beta[c]);
    }
}

// ---------------------------------------------------------------------------
// Kernel 2: full slot-attention iteration loop. One block per (b,t) frame.
// 56 KB static LDS arena, 8 waves, all GEMMs via v_wmma_f32_16x16x32_bf16.
// ---------------------------------------------------------------------------
__global__ __launch_bounds__(256)
void slot_kernel(const float* __restrict__ noise,
                 const float* __restrict__ mu, const float* __restrict__ sigma,
                 const float* __restrict__ ln_s_g, const float* __restrict__ ln_s_b,
                 const float* __restrict__ ln_m_g, const float* __restrict__ ln_m_b,
                 const float* __restrict__ bq,
                 const float* __restrict__ b_ih, const float* __restrict__ b_hh,
                 const float* __restrict__ b1, const float* __restrict__ b2,
                 const float* __restrict__ Wpos, const float* __restrict__ bpos,
                 const float* __restrict__ Wex, const float* __restrict__ bex,
                 const bf16_t* __restrict__ WqT,
                 const bf16_t* __restrict__ Wih, const bf16_t* __restrict__ Whh,
                 const bf16_t* __restrict__ W1T, const bf16_t* __restrict__ W2T,
                 const bf16_t* __restrict__ kbf, const bf16_t* __restrict__ vT,
                 float* __restrict__ out_slots, float* __restrict__ out_pos,
                 float* __restrict__ out_ex) {
    // 56 KB arena with phase-based reuse
    __shared__ __align__(32) char arena[57344];
    float*  s_slots = (float*)(arena);            // 16 KB persistent slot state (fp32)
    bf16_t* s_prev  = (bf16_t*)(arena + 16384);   //  8 KB prev slots (bf16, GRU A + blend)
    bf16_t* s_bufA  = (bf16_t*)(arena + 24576);   // 16 KB attn chunk [16][512] / mlp1 [16][512]
    bf16_t* s_bufB  = (bf16_t*)(arena + 40960);   //  8 KB lnS / upd / ln_m  [16][256]
    bf16_t* s_q     = (bf16_t*)(arena + 49152);   //  8 KB q [16][256]

    const int tid     = threadIdx.x;
    const int wave    = tid >> 5;
    const int lane    = tid & 31;
    const int halfsel = lane >> 4;
    const int col16   = lane & 15;
    const int bt      = blockIdx.x;

    // slots = mu + sigma * noise
    for (int i = tid; i < 4096; i += 256)
        s_slots[i] = mu[i & 255] + sigma[i & 255] * noise[(size_t)bt * 4096 + i];
    __syncthreads();

    const bf16_t* kb = kbf + (size_t)bt * 1024 * 256;  // [n][d] bf16
    const bf16_t* vb = vT  + (size_t)bt * 256 * 1024;  // [d][n] bf16
    const float scale = 0.0625f;                       // 256^-0.5

    for (int it = 0; it < 3; ++it) {
        // ---- prev snapshot + LN_s(slots) -> s_bufB ----
        for (int i = tid; i < 4096; i += 256) s_prev[i] = (bf16_t)s_slots[i];
        __syncthreads();
        ln16(s_slots, ln_s_g, ln_s_b, s_bufB);
        __syncthreads();

        // ---- q = lnS @ WqT + bq -> s_q (bf16, row-major [s][d]) ----
        #pragma unroll
        for (int jj = 0; jj < 2; ++jj) {
            int nt = wave + 8 * jj;
            f32x8 acc = {};
            for (int k0 = 0; k0 < 256; k0 += 32) {
                bf16x16 a = load_a16(s_bufB + k0, 256);
                bf16x16 b = load_b16(WqT + (size_t)nt * 16 * 256 + k0, 256);
                acc = WMMA_BF16(a, b, acc);
            }
            int c = nt * 16 + col16;
            #pragma unroll
            for (int r = 0; r < 8; ++r)
                s_q[(r + halfsel * 8) * 256 + c] = (bf16_t)(acc[r] + bq[c]);
        }
        __syncthreads();

        // ---- attn logits (attnT = k @ qT), softmax over slots, upd += attn @ v ----
        f32x8 uacc0 = {}, uacc1 = {};
        for (int ch = 0; ch < 2; ++ch) {
            // attnT for tokens n in [ch*512, ch*512+512): 32 m-tiles, 4 per wave
            #pragma unroll
            for (int jj = 0; jj < 4; ++jj) {
                int mt = wave + 8 * jj;
                f32x8 acc = {};
                const bf16_t* arow = kb + (size_t)(ch * 512 + mt * 16) * 256;
                for (int k0 = 0; k0 < 256; k0 += 32) {
                    bf16x16 a = load_a16(arow + k0, 256);
                    bf16x16 b = load_b16(s_q + k0, 256);
                    acc = WMMA_BF16(a, b, acc);
                }
                // softmax over slots: the 16 slot values for one token live in the
                // 16 lanes of this half-wave -> shfl_xor reduction (wave32-native)
                #pragma unroll
                for (int r = 0; r < 8; ++r) {
                    float v = acc[r] * scale;
                    float m = v;
                    m = fmaxf(m, __shfl_xor(m, 1, 32));
                    m = fmaxf(m, __shfl_xor(m, 2, 32));
                    m = fmaxf(m, __shfl_xor(m, 4, 32));
                    m = fmaxf(m, __shfl_xor(m, 8, 32));
                    float e = __expf(v - m);
                    float s = e;
                    s += __shfl_xor(s, 1, 32);
                    s += __shfl_xor(s, 2, 32);
                    s += __shfl_xor(s, 4, 32);
                    s += __shfl_xor(s, 8, 32);
                    int n_loc = mt * 16 + r + halfsel * 8;
                    s_bufA[col16 * 512 + n_loc] = (bf16_t)(e / s);  // attn[s][n] bf16
                }
            }
            __syncthreads();
            // partial upd: 2 d-tiles per wave, accumulators persist across chunks
            for (int kk = 0; kk < 512; kk += 32) {
                bf16x16 a = load_a16(s_bufA + kk, 512);
                bf16x16 b0 = load_b16(vb + (size_t)(wave * 2 + 0) * 16 * 1024 + ch * 512 + kk, 1024);
                bf16x16 b1v = load_b16(vb + (size_t)(wave * 2 + 1) * 16 * 1024 + ch * 512 + kk, 1024);
                uacc0 = WMMA_BF16(a, b0, uacc0);
                uacc1 = WMMA_BF16(a, b1v, uacc1);
            }
            __syncthreads();
        }
        // upd -> s_bufB (bf16, [s][d])
        {
            int d0 = (wave * 2 + 0) * 16 + col16;
            int d1 = (wave * 2 + 1) * 16 + col16;
            #pragma unroll
            for (int r = 0; r < 8; ++r) {
                s_bufB[(r + halfsel * 8) * 256 + d0] = (bf16_t)uacc0[r];
                s_bufB[(r + halfsel * 8) * 256 + d1] = (bf16_t)uacc1[r];
            }
        }
        __syncthreads();

        // ---- GRU: per column-tile, compute all 6 gate GEMMs in registers ----
        #pragma unroll
        for (int jj = 0; jj < 2; ++jj) {
            int t = wave + 8 * jj;
            f32x8 a_ir = {}, a_iz = {}, a_in = {}, a_hr = {}, a_hz = {}, a_hn = {};
            for (int k0 = 0; k0 < 256; k0 += 32) {
                bf16x16 au = load_a16(s_bufB + k0, 256);
                bf16x16 ap = load_a16(s_prev + k0, 256);
                bf16x16 bir = load_b16(Wih + (size_t)(t     ) * 16 * 256 + k0, 256);
                bf16x16 biz = load_b16(Wih + (size_t)(t + 16) * 16 * 256 + k0, 256);
                bf16x16 bin = load_b16(Wih + (size_t)(t + 32) * 16 * 256 + k0, 256);
                bf16x16 bhr = load_b16(Whh + (size_t)(t     ) * 16 * 256 + k0, 256);
                bf16x16 bhz = load_b16(Whh + (size_t)(t + 16) * 16 * 256 + k0, 256);
                bf16x16 bhn = load_b16(Whh + (size_t)(t + 32) * 16 * 256 + k0, 256);
                a_ir = WMMA_BF16(au, bir, a_ir);
                a_iz = WMMA_BF16(au, biz, a_iz);
                a_in = WMMA_BF16(au, bin, a_in);
                a_hr = WMMA_BF16(ap, bhr, a_hr);
                a_hz = WMMA_BF16(ap, bhz, a_hz);
                a_hn = WMMA_BF16(ap, bhn, a_hn);
            }
            int c = t * 16 + col16;
            #pragma unroll
            for (int r = 0; r < 8; ++r) {
                int row = r + halfsel * 8;
                float ir = a_ir[r] + b_ih[c];
                float hr = a_hr[r] + b_hh[c];
                float iz = a_iz[r] + b_ih[c + 256];
                float hz = a_hz[r] + b_hh[c + 256];
                float nn = a_in[r] + b_ih[c + 512];
                float hn = a_hn[r] + b_hh[c + 512];
                float rg = sigmoidf_(ir + hr);
                float zg = sigmoidf_(iz + hz);
                float ng = tanhf(nn + rg * hn);
                float pv = (float)s_prev[row * 256 + c];
                s_slots[row * 256 + c] = (1.f - zg) * ng + zg * pv;
            }
        }
        __syncthreads();

        // ---- MLP residual: slots += relu(ln_m(slots) @ W1 + b1) @ W2 + b2 ----
        ln16(s_slots, ln_m_g, ln_m_b, s_bufB);
        __syncthreads();
        #pragma unroll
        for (int jj = 0; jj < 4; ++jj) {
            int nt = wave + 8 * jj;
            f32x8 acc = {};
            for (int k0 = 0; k0 < 256; k0 += 32) {
                bf16x16 a = load_a16(s_bufB + k0, 256);
                bf16x16 b = load_b16(W1T + (size_t)nt * 16 * 256 + k0, 256);
                acc = WMMA_BF16(a, b, acc);
            }
            int c = nt * 16 + col16;
            #pragma unroll
            for (int r = 0; r < 8; ++r)
                s_bufA[(r + halfsel * 8) * 512 + c] = (bf16_t)fmaxf(acc[r] + b1[c], 0.f);
        }
        __syncthreads();
        #pragma unroll
        for (int jj = 0; jj < 2; ++jj) {
            int nt = wave + 8 * jj;
            f32x8 acc = {};
            for (int k0 = 0; k0 < 512; k0 += 32) {
                bf16x16 a = load_a16(s_bufA + k0, 512);
                bf16x16 b = load_b16(W2T + (size_t)nt * 16 * 512 + k0, 512);
                acc = WMMA_BF16(a, b, acc);
            }
            int c = nt * 16 + col16;
            #pragma unroll
            for (int r = 0; r < 8; ++r)
                s_slots[(r + halfsel * 8) * 256 + c] += acc[r] + b2[c];
        }
        __syncthreads();
    }

    // ---- outputs ----
    for (int i = tid; i < 4096; i += 256)
        out_slots[(size_t)bt * 4096 + i] = s_slots[i];
    if (tid < 64) {
        int s = tid >> 2, j = tid & 3;
        float acc = bpos[j];
        for (int d = 0; d < 256; ++d) acc += s_slots[s * 256 + d] * Wpos[d * 4 + j];
        out_pos[bt * 64 + s * 4 + j] = acc;
    } else if (tid < 80) {
        int s = tid - 64;
        float acc = bex[0];
        for (int d = 0; d < 256; ++d) acc += s_slots[s * 256 + d] * Wex[d];
        out_ex[bt * 16 + s] = sigmoidf_(acc);
    }
}

// ---------------------------------------------------------------------------
// Host launcher
// ---------------------------------------------------------------------------
extern "C" void kernel_launch(void* const* d_in, const int* in_sizes, int n_in,
                              void* d_out, int out_size, void* d_ws, size_t ws_size,
                              hipStream_t stream) {
    (void)in_sizes; (void)n_in; (void)out_size; (void)ws_size;

    const float* features = (const float*)d_in[0];
    const float* noise    = (const float*)d_in[1];
    const float* ln_in_g  = (const float*)d_in[2];
    const float* ln_in_b  = (const float*)d_in[3];
    const float* ln_s_g   = (const float*)d_in[4];
    const float* ln_s_b   = (const float*)d_in[5];
    const float* ln_m_g   = (const float*)d_in[6];
    const float* ln_m_b   = (const float*)d_in[7];
    const float* mu       = (const float*)d_in[8];
    const float* sigma    = (const float*)d_in[9];
    const float* Wk       = (const float*)d_in[10];
    const float* bk       = (const float*)d_in[11];
    const float* Wv       = (const float*)d_in[12];
    const float* bv       = (const float*)d_in[13];
    const float* Wq       = (const float*)d_in[14];
    const float* bq       = (const float*)d_in[15];
    const float* W_ih     = (const float*)d_in[16];
    const float* b_ih     = (const float*)d_in[17];
    const float* W_hh     = (const float*)d_in[18];
    const float* b_hh     = (const float*)d_in[19];
    const float* W1       = (const float*)d_in[20];
    const float* b1       = (const float*)d_in[21];
    const float* W2       = (const float*)d_in[22];
    const float* b2       = (const float*)d_in[23];
    const float* Wpos     = (const float*)d_in[24];
    const float* bpos     = (const float*)d_in[25];
    const float* Wex      = (const float*)d_in[26];
    const float* bex      = (const float*)d_in[27];

    // Workspace layout (bytes)
    char* ws = (char*)d_ws;
    bf16_t* WkvT = (bf16_t*)(ws + 0);         // [512][512]
    bf16_t* WqT  = (bf16_t*)(ws + 524288);    // [256][256]
    bf16_t* Wihb = (bf16_t*)(ws + 655360);    // [768][256] (already Bt layout)
    bf16_t* Whhb = (bf16_t*)(ws + 1048576);   // [768][256]
    bf16_t* W1T  = (bf16_t*)(ws + 1441792);   // [512][256]
    bf16_t* W2T  = (bf16_t*)(ws + 1703936);   // [256][512]
    bf16_t* kbf  = (bf16_t*)(ws + 1966080);   // [65536][256] bf16 (32 MB)
    bf16_t* vTp  = (bf16_t*)(ws + 35520512);  // [64][256][1024] bf16 (32 MB)

    pack_bt<<<64, 256, 0, stream>>>(Wk,   WkvT,             512, 256, 1);
    pack_bt<<<64, 256, 0, stream>>>(Wv,   WkvT + 256 * 512, 512, 256, 1);
    pack_bt<<<64, 256, 0, stream>>>(Wq,   WqT,              256, 256, 1);
    pack_bt<<<64, 256, 0, stream>>>(W_ih, Wihb,             256, 768, 0);
    pack_bt<<<64, 256, 0, stream>>>(W_hh, Whhb,             256, 768, 0);
    pack_bt<<<64, 256, 0, stream>>>(W1,   W1T,              256, 512, 1);
    pack_bt<<<64, 256, 0, stream>>>(W2,   W2T,              512, 256, 1);

    ln_kv_kernel<<<4096, 256, 0, stream>>>(features, ln_in_g, ln_in_b,
                                           WkvT, bk, bv, kbf, vTp);

    float* out = (float*)d_out;
    slot_kernel<<<64, 256, 0, stream>>>(noise, mu, sigma,
                                        ln_s_g, ln_s_b, ln_m_g, ln_m_b,
                                        bq, b_ih, b_hh, b1, b2,
                                        Wpos, bpos, Wex, bex,
                                        WqT, Wihb, Whhb, W1T, W2T,
                                        kbf, vTp,
                                        out, out + 262144, out + 262144 + 4096);
}
